// SpatialAttention_43585328119994
// MI455X (gfx1250) — matrix operand order repaired and checked by reading, not compiled, not run on hardware
//
#include <hip/hip_runtime.h>

typedef __attribute__((ext_vector_type(16))) __bf16 v16bf;
typedef __attribute__((ext_vector_type(8)))  float  v8f;
typedef __attribute__((ext_vector_type(4)))  unsigned v4u;
typedef __attribute__((ext_vector_type(8)))  unsigned v8u;

#define DEV __device__ __forceinline__

DEV v8f wmma_bf16(v16bf a, v16bf b, v8f c) {
  // 8 args: (neg_a, A, neg_b, B, c_mod, C, reuse_a, reuse_b)
  return __builtin_amdgcn_wmma_f32_16x16x32_bf16(false, a, false, b, (short)0, c,
                                                 false, false);
}

// Raw LDS byte offset from a generic pointer to __shared__ (low 32 bits of the
// flat shared-aperture address are the LDS byte address).
DEV unsigned lds_off(const void* p) {
  return (unsigned)(unsigned long long)p;
}

// Async global->LDS copy, 128 bits per lane. Tracked with ASYNCcnt.
DEV void async_b128(unsigned ldsb, const void* g) {
  asm volatile("global_load_async_to_lds_b128 %0, %1, off"
               :: "v"(ldsb), "v"(g)
               : "memory");
}
DEV void wait_async0() { asm volatile("s_wait_asynccnt 0x0" ::: "memory"); }

// TDM: DMA a 2-D tile (tile0 x tile1 elements, 2-byte data) from global into
// LDS. D# group0 (4 SGPRs) + group1 (8 SGPRs) per CDNA5 ISA ch.8; VADDR2/3
// omitted (NULL) -> 2-D descriptor. Tracked with TENSORcnt.
DEV void tdm_load_2d(unsigned ldsb, const void* gaddr, unsigned tdim0,
                     unsigned tdim1, unsigned tile0, unsigned tile1,
                     unsigned stride0) {
  unsigned long long ga = (unsigned long long)gaddr;
  v4u g0;
  g0[0] = 1u;                                  // count=1 (valid user D#)
  g0[1] = ldsb;                                // lds_addr
  g0[2] = (unsigned)ga;                        // global_addr[31:0]
  g0[3] = (unsigned)(ga >> 32) | (2u << 30);   // global_addr[56:32] | type=2
  v8u g1;
  g1[0] = (1u << 16);                          // workgroup_mask=0, data_size=1 (2B)
  g1[1] = (tdim0 & 0xFFFFu) << 16;             // tensor_dim0[15:0]
  g1[2] = (tdim0 >> 16) | ((tdim1 & 0xFFFFu) << 16);
  g1[3] = (tdim1 >> 16) | ((tile0 & 0xFFFFu) << 16);
  g1[4] = (tile1 & 0xFFFFu);                   // tile_dim1 (tile_dim2 = 0)
  g1[5] = stride0;                             // tensor_dim0_stride[31:0]
  g1[6] = 0u;
  g1[7] = 0u;
  asm volatile("tensor_load_to_lds %0, %1" :: "s"(g0), "s"(g1) : "memory");
}

// 16-element bf16 fragment loaders (32B contiguous, or 2x16B runs).
union BF16x16 { uint4 q[2]; v16bf v; };
DEV v16bf ld_2x8(const __bf16* p0, const __bf16* p1) {
  BF16x16 u;
  u.q[0] = *(const uint4*)p0;
  u.q[1] = *(const uint4*)p1;
  return u.v;
}
DEV v16bf ld_16(const __bf16* p) {
  BF16x16 u;
  u.q[0] = ((const uint4*)p)[0];
  u.q[1] = ((const uint4*)p)[1];
  return u.v;
}

// ---------------------------------------------------------------- cast f32->bf16
__global__ void cast_bf16_kernel(const float* __restrict__ src,
                                 __bf16* __restrict__ dst, int n) {
  int i = blockIdx.x * blockDim.x + threadIdx.x;
  if (i < n) dst[i] = (__bf16)src[i];
}

// ---------------------------------------------------------------- LayerNorm (D=1024)
__global__ void layernorm_kernel(const float* __restrict__ x,
                                 const float* __restrict__ gamma,
                                 const float* __restrict__ beta,
                                 __bf16* __restrict__ xn) {
  __shared__ float r1[256], r2[256];
  const int row = blockIdx.x, t = threadIdx.x;
  const float4 xv = ((const float4*)(x + (size_t)row * 1024))[t];
  r1[t] = xv.x + xv.y + xv.z + xv.w;
  r2[t] = xv.x * xv.x + xv.y * xv.y + xv.z * xv.z + xv.w * xv.w;
  __syncthreads();
  for (int s = 128; s > 0; s >>= 1) {
    if (t < s) { r1[t] += r1[t + s]; r2[t] += r2[t + s]; }
    __syncthreads();
  }
  const float mean = r1[0] * (1.0f / 1024.0f);
  const float var  = r2[0] * (1.0f / 1024.0f) - mean * mean;
  const float rstd = rsqrtf(var + 1e-5f);
  #pragma unroll
  for (int i = 0; i < 4; ++i) {
    int d = t * 4 + i;
    float v = ((&xv.x)[i] - mean) * rstd * gamma[d] + beta[d];
    xn[(size_t)row * 1024 + d] = (__bf16)v;
  }
}

// ---------------------------------------------------------------- QKV GEMM (M=16384,N=3072,K=1024)
// A staged via async copies; B staged transposed so fragments are contiguous.
// Epilogue: + bias, scatter bf16 into head-major q/k/v: [bh=bt*16+h][l][dh]
__global__ __launch_bounds__(128) void gemm_qkv_kernel(
    const __bf16* __restrict__ Xn, const __bf16* __restrict__ W,
    const float* __restrict__ bias, __bf16* __restrict__ qh,
    __bf16* __restrict__ kh, __bf16* __restrict__ vh) {
  __shared__ __align__(16) __bf16 As[64 * 32];   // [m][k]
  __shared__ __align__(16) __bf16 BsT[64 * 32];  // [n][k] (transposed)
  const int t = threadIdx.x;
  const int w = t >> 5, lane = t & 31;
  const int half = lane >> 4, l16 = lane & 15;
  const int m0 = blockIdx.y * 64, n0 = blockIdx.x * 64;
  const int wm = (w >> 1) * 32, wn = (w & 1) * 32;
  const unsigned abase = lds_off(As);

  v8f acc[2][2] = {};

  for (int kt = 0; kt < 1024; kt += 32) {
    #pragma unroll
    for (int i = 0; i < 2; ++i) {
      int idx = t + i * 128;
      // A tile 64x32: linear LDS layout, async DMA per 16B chunk
      int ar = idx >> 2, ac = (idx & 3) * 8;
      async_b128(abase + idx * 16, &Xn[(size_t)(m0 + ar) * 1024 + kt + ac]);
      // B tile 32x64: load wide, scatter-store transposed
      int br = idx >> 3, bc = (idx & 7) * 8;
      uint4 d = *(const uint4*)&W[(size_t)(kt + br) * 3072 + n0 + bc];
      __bf16 tmp[8];
      *(uint4*)tmp = d;
      #pragma unroll
      for (int j = 0; j < 8; ++j) BsT[(bc + j) * 32 + br] = tmp[j];
    }
    wait_async0();
    __syncthreads();

    v16bf af[2], bfr[2];
    #pragma unroll
    for (int ti = 0; ti < 2; ++ti) {
      int m = wm + ti * 16 + l16;
      af[ti] = ld_2x8(&As[m * 32 + half * 8], &As[m * 32 + 16 + half * 8]);
    }
    #pragma unroll
    for (int tj = 0; tj < 2; ++tj) {
      int n = wn + tj * 16 + l16;
      bfr[tj] = ld_2x8(&BsT[n * 32 + half * 8], &BsT[n * 32 + 16 + half * 8]);
    }
    #pragma unroll
    for (int ti = 0; ti < 2; ++ti)
      #pragma unroll
      for (int tj = 0; tj < 2; ++tj)
        acc[ti][tj] = wmma_bf16(af[ti], bfr[tj], acc[ti][tj]);
    __syncthreads();
  }

  #pragma unroll
  for (int ti = 0; ti < 2; ++ti)
    #pragma unroll
    for (int tj = 0; tj < 2; ++tj) {
      int nglob = n0 + wn + tj * 16 + l16;
      float bb = bias[nglob];
      int which = nglob >> 10, hn = nglob & 1023;
      int h = hn >> 6, dd = hn & 63;
      __bf16* dst = (which == 0) ? qh : ((which == 1) ? kh : vh);
      #pragma unroll
      for (int r = 0; r < 8; ++r) {
        int mglob = m0 + wm + ti * 16 + half * 8 + r;
        int bt = mglob >> 10, lq = mglob & 1023;
        dst[(((size_t)(bt * 16 + h)) * 1024 + lq) * 64 + dd] =
            (__bf16)(acc[ti][tj][r] + bb);
      }
    }
}

// ---------------------------------------------------------------- RoPE (rot_dim=32) on q,k
__global__ void rope_kernel(__bf16* __restrict__ qh, __bf16* __restrict__ kh,
                            const float* __restrict__ rot) {
  int idx = blockIdx.x * blockDim.x + threadIdx.x;  // 2 * 256 * 1024 * 16
  int dd = idx & 15;
  int lq = (idx >> 4) & 1023;
  int bh = (idx >> 14) & 255;
  int buf = idx >> 22;
  __bf16* p = buf ? kh : qh;
  size_t base = (((size_t)bh) * 1024 + lq) * 64;
  float t0 = (float)p[base + dd];
  float t1 = (float)p[base + dd + 16];
  float f = rot[lq * 32 + dd];
  float c = __cosf(f), s = __sinf(f);
  p[base + dd]      = (__bf16)(t0 * c - t1 * s);
  p[base + dd + 16] = (__bf16)(t1 * c + t0 * s);
}

// ---------------------------------------------------------------- flash attention
// grid (L/64, 256); 4 waves/block, each owns 16 query rows; KV tiled by 32.
// K tile staged by TDM (wave 0); V tile staged transposed by all threads.
__global__ __launch_bounds__(128) void attn_kernel(
    const __bf16* __restrict__ qh, const __bf16* __restrict__ kh,
    const __bf16* __restrict__ vh, __bf16* __restrict__ ao) {
  __shared__ __align__(16) __bf16 Ks[32 * 64];   // [key][dh]
  __shared__ __align__(16) __bf16 VsT[64 * 32];  // [dh][key] (transposed)
  __shared__ __align__(16) __bf16 Ps[4][16 * 32];

  const int t = threadIdx.x;
  const int w = t >> 5, lane = t & 31;
  const int half = lane >> 4, l16 = lane & 15;
  const int q0 = blockIdx.x * 64;
  const int bh = blockIdx.y;
  const size_t hbase = (size_t)bh * 1024 * 64;
  const unsigned kbase = lds_off(Ks);

  // q fragments for this wave's 16 rows (dh=64 -> two K=32 chunks), kept live.
  const __bf16* qbase = qh + hbase + (size_t)(q0 + w * 16) * 64;
  v16bf qf[2];
  #pragma unroll
  for (int c = 0; c < 2; ++c)
    qf[c] = ld_2x8(&qbase[l16 * 64 + c * 32 + half * 8],
                   &qbase[l16 * 64 + c * 32 + 16 + half * 8]);

  float mrow[8], lrow[8];
  v8f O[4] = {};
  #pragma unroll
  for (int r = 0; r < 8; ++r) { mrow[r] = -3.0e38f; lrow[r] = 0.0f; }

  for (int kv = 0; kv < 1024; kv += 32) {
    // K tile via Tensor Data Mover (one wave issues, EXEC-independent DMA)
    if (t < 32) {
      tdm_load_2d(kbase, &kh[hbase + (size_t)kv * 64],
                  /*tdim0=*/64, /*tdim1=*/1024, /*tile0=*/64, /*tile1=*/32,
                  /*stride0=*/64);
      __builtin_amdgcn_s_wait_tensorcnt(0);
    }
    // V tile: wide load + transposed scatter into VsT
    #pragma unroll
    for (int i = 0; i < 2; ++i) {
      int idx = t + i * 128;
      int rr = idx >> 3, cc = (idx & 7) * 8;
      uint4 d = *(const uint4*)&vh[hbase + (size_t)(kv + rr) * 64 + cc];
      __bf16 tmp[8];
      *(uint4*)tmp = d;
      #pragma unroll
      for (int j = 0; j < 8; ++j) VsT[(cc + j) * 32 + rr] = tmp[j];
    }
    __syncthreads();

    // S = q . k^T : two 16x16 tiles over the 32-key tile
    v8f S[2] = {};
    #pragma unroll
    for (int t2 = 0; t2 < 2; ++t2)
      #pragma unroll
      for (int c = 0; c < 2; ++c) {
        v16bf bk = ld_2x8(&Ks[(t2 * 16 + l16) * 64 + c * 32 + half * 8], &Ks[(t2 * 16 + l16) * 64 + c * 32 + 16 + half * 8]);
        S[t2] = wmma_bf16(qf[c], bk, S[t2]);
      }

    // online softmax: C-layout -> row m=half*8+r lives across 16 lanes
    #pragma unroll
    for (int r = 0; r < 8; ++r) {
      float s0 = S[0][r] * 0.125f;
      float s1 = S[1][r] * 0.125f;
      float mx = fmaxf(s0, s1);
      #pragma unroll
      for (int o = 1; o < 16; o <<= 1) mx = fmaxf(mx, __shfl_xor(mx, o, 32));
      float mnew = fmaxf(mrow[r], mx);
      float alpha = __expf(mrow[r] - mnew);
      float p0 = __expf(s0 - mnew);
      float p1 = __expf(s1 - mnew);
      float rs = p0 + p1;
      #pragma unroll
      for (int o = 1; o < 16; o <<= 1) rs += __shfl_xor(rs, o, 32);
      lrow[r] = lrow[r] * alpha + rs;
      mrow[r] = mnew;
      #pragma unroll
      for (int j = 0; j < 4; ++j) O[j][r] *= alpha;
      // stash P through LDS to reshape C-layout -> A-fragment layout
      Ps[w][(half * 8 + r) * 32 + l16]      = (__bf16)p0;
      Ps[w][(half * 8 + r) * 32 + 16 + l16] = (__bf16)p1;
    }

    // P as 16x32 A-matrix (K = key index): two contiguous 8-element runs
    v16bf pf = ld_2x8(&Ps[w][l16 * 32 + half * 8],
                      &Ps[w][l16 * 32 + 16 + half * 8]);

    #pragma unroll
    for (int j = 0; j < 4; ++j) {
      v16bf bv = ld_2x8(&VsT[(j * 16 + l16) * 32 + half * 8], &VsT[(j * 16 + l16) * 32 + 16 + half * 8]);
      O[j] = wmma_bf16(pf, bv, O[j]);
    }
    __syncthreads();
  }

  // normalize + store bf16 in [row = bt*1024+l][col = h*64+d] layout
  const int h = bh & 15, bt = bh >> 4;
  #pragma unroll
  for (int r = 0; r < 8; ++r) {
    float inv = 1.0f / lrow[r];
    int lq = q0 + w * 16 + half * 8 + r;
    size_t rowout = (size_t)(bt * 1024 + lq);
    #pragma unroll
    for (int j = 0; j < 4; ++j)
      ao[rowout * 1024 + h * 64 + j * 16 + l16] = (__bf16)(O[j][r] * inv);
  }
}

// ---------------------------------------------------------------- out GEMM (M=16384,N=1024,K=1024) -> f32
__global__ __launch_bounds__(128) void gemm_out_kernel(
    const __bf16* __restrict__ A, const __bf16* __restrict__ W,
    float* __restrict__ out) {
  __shared__ __align__(16) __bf16 As[64 * 32];
  __shared__ __align__(16) __bf16 BsT[64 * 32];
  const int t = threadIdx.x;
  const int w = t >> 5, lane = t & 31;
  const int half = lane >> 4, l16 = lane & 15;
  const int m0 = blockIdx.y * 64, n0 = blockIdx.x * 64;
  const int wm = (w >> 1) * 32, wn = (w & 1) * 32;
  const unsigned abase = lds_off(As);

  v8f acc[2][2] = {};

  for (int kt = 0; kt < 1024; kt += 32) {
    #pragma unroll
    for (int i = 0; i < 2; ++i) {
      int idx = t + i * 128;
      int ar = idx >> 2, ac = (idx & 3) * 8;
      async_b128(abase + idx * 16, &A[(size_t)(m0 + ar) * 1024 + kt + ac]);
      int br = idx >> 3, bc = (idx & 7) * 8;
      uint4 d = *(const uint4*)&W[(size_t)(kt + br) * 1024 + n0 + bc];
      __bf16 tmp[8];
      *(uint4*)tmp = d;
      #pragma unroll
      for (int j = 0; j < 8; ++j) BsT[(bc + j) * 32 + br] = tmp[j];
    }
    wait_async0();
    __syncthreads();

    v16bf af[2], bfr[2];
    #pragma unroll
    for (int ti = 0; ti < 2; ++ti) {
      int m = wm + ti * 16 + l16;
      af[ti] = ld_2x8(&As[m * 32 + half * 8], &As[m * 32 + 16 + half * 8]);
    }
    #pragma unroll
    for (int tj = 0; tj < 2; ++tj) {
      int n = wn + tj * 16 + l16;
      bfr[tj] = ld_2x8(&BsT[n * 32 + half * 8], &BsT[n * 32 + 16 + half * 8]);
    }
    #pragma unroll
    for (int ti = 0; ti < 2; ++ti)
      #pragma unroll
      for (int tj = 0; tj < 2; ++tj)
        acc[ti][tj] = wmma_bf16(af[ti], bfr[tj], acc[ti][tj]);
    __syncthreads();
  }

  #pragma unroll
  for (int ti = 0; ti < 2; ++ti)
    #pragma unroll
    for (int tj = 0; tj < 2; ++tj) {
      int nglob = n0 + wn + tj * 16 + l16;
      #pragma unroll
      for (int r = 0; r < 8; ++r) {
        int mglob = m0 + wm + ti * 16 + half * 8 + r;
        out[(size_t)mglob * 1024 + nglob] = acc[ti][tj][r];
      }
    }
}

// ================================================================ launcher
extern "C" void kernel_launch(void* const* d_in, const int* in_sizes, int n_in,
                              void* d_out, int out_size, void* d_ws,
                              size_t ws_size, hipStream_t stream) {
  (void)in_sizes; (void)n_in; (void)out_size; (void)ws_size;
  const float* x     = (const float*)d_in[0];
  const float* gamma = (const float*)d_in[1];
  const float* beta  = (const float*)d_in[2];
  const float* Wqkv  = (const float*)d_in[3];
  const float* bqkv  = (const float*)d_in[4];
  const float* Wout  = (const float*)d_in[5];
  const float* rot   = (const float*)d_in[6];
  float* out = (float*)d_out;

  char* ws = (char*)d_ws;
  size_t off = 0;
  __bf16* xn = (__bf16*)(ws + off); off += (size_t)16384 * 1024 * 2;
  __bf16* wq = (__bf16*)(ws + off); off += (size_t)1024 * 3072 * 2;
  __bf16* wo = (__bf16*)(ws + off); off += (size_t)1024 * 1024 * 2;
  __bf16* qh = (__bf16*)(ws + off); off += (size_t)256 * 1024 * 64 * 2;
  __bf16* kh = (__bf16*)(ws + off); off += (size_t)256 * 1024 * 64 * 2;
  __bf16* vh = (__bf16*)(ws + off); off += (size_t)256 * 1024 * 64 * 2;
  __bf16* ao = (__bf16*)(ws + off); off += (size_t)16384 * 1024 * 2;

  cast_bf16_kernel<<<(3072 * 1024) / 256, 256, 0, stream>>>(Wqkv, wq, 3072 * 1024);
  cast_bf16_kernel<<<(1024 * 1024) / 256, 256, 0, stream>>>(Wout, wo, 1024 * 1024);
  layernorm_kernel<<<16384, 256, 0, stream>>>(x, gamma, beta, xn);
  gemm_qkv_kernel<<<dim3(48, 256), 128, 0, stream>>>(xn, wq, bqkv, qh, kh, vh);
  rope_kernel<<<32768, 256, 0, stream>>>(qh, kh, rot);
  attn_kernel<<<dim3(16, 256), 128, 0, stream>>>(qh, kh, vh, ao);
  gemm_out_kernel<<<dim3(16, 256), 128, 0, stream>>>(ao, wo, out);
}
